// MultiHeadSelfAttention_44753559224707
// MI455X (gfx1250) — matrix operand order repaired and checked by reading, not compiled, not run on hardware
//
#include <hip/hip_runtime.h>
#include <stdint.h>

typedef unsigned short u16;
typedef __attribute__((ext_vector_type(16))) __bf16 v16bf;
typedef __attribute__((ext_vector_type(8)))  float  v8f;
typedef __attribute__((ext_vector_type(4)))  int    v4i;

#define N_EMBD 1024
#define N_HEAD 16
#define DH     64
#define TSEQ   2048
#define BATCH  4
#define MROWS  (BATCH * TSEQ)   /* 8192 */
#define QKV_N  (3 * N_EMBD)     /* 3072 */
#define INV_SCALE 8.0f          /* sqrt(Dh): faithful to the reference's /scale bug */

#if defined(__has_builtin)
#if __has_builtin(__builtin_amdgcn_global_load_async_to_lds_b128)
#define HAVE_ASYNC_LDS 1
#endif
#endif
#ifndef HAVE_ASYNC_LDS
#define HAVE_ASYNC_LDS 0
#endif

typedef __attribute__((address_space(1))) void gvoid;
typedef __attribute__((address_space(3))) void lvoid;
typedef __attribute__((address_space(1))) v4i  gv4i;
typedef __attribute__((address_space(3))) v4i  lv4i;

#define AS_GLOBAL_V4I(p) ((gv4i*)(gvoid*)(void*)(p))
#define AS_LOCAL_V4I(p)  ((lv4i*)(lvoid*)(void*)(p))

__device__ __forceinline__ void wait_asynccnt_0() {
#if __has_builtin(__builtin_amdgcn_s_wait_asynccnt)
  __builtin_amdgcn_s_wait_asynccnt(0);
#else
  asm volatile("s_wait_asynccnt 0x0" ::: "memory");
#endif
}

__device__ __forceinline__ u16 f2bf(float f) {
  union { float f; unsigned u; } c; c.f = f;
  unsigned u = c.u;
  u += 0x7FFFu + ((u >> 16) & 1u);   // round-to-nearest-even
  return (u16)(u >> 16);
}

union Frag16 { v16bf v; uint4 q[2]; };

// A-matrix 16x32 bf16 fragment (ISA 7.12.2): lane holds row r0+(lane&15);
// elems 0..7 = K kb+off.. , elems 8..15 = K kb+off+16.. ; off = (lane>=16)?8:0
__device__ __forceinline__ v16bf load_frag_a(const u16* base, int r0, int ld, int kb, int lane) {
  const u16* p = base + (size_t)(r0 + (lane & 15)) * ld + kb + ((lane >> 4) << 3);
  Frag16 f;
  f.q[0] = *(const uint4*)p;
  f.q[1] = *(const uint4*)(p + 16);
  return f.v;
}

// B-matrix 32x16 bf16 fragment: lane holds column c0+(lane&15);
// elems 0..7 = K kb+off.. , elems 8..15 = K kb+off+16.. ; off = (lane>=16)?8:0
__device__ __forceinline__ v16bf load_frag_b(const u16* base, int c0, int ld, int kb, int lane) {
  const u16* p = base + (size_t)(c0 + (lane & 15)) * ld + kb + ((lane >> 4) << 3);
  Frag16 f;
  f.q[0] = *(const uint4*)p;
  f.q[1] = *(const uint4*)(p + 16);
  return f.v;
}

__device__ __forceinline__ v8f wmma_bf16(v16bf a, v16bf b, v8f c) {
  return __builtin_amdgcn_wmma_f32_16x16x32_bf16(false, a, false, b, (short)0, c, false, false);
}

// ---------------- conversion kernels ----------------

__global__ void cvt_bf16x4_kernel(const float4* __restrict__ in, ushort4* __restrict__ out, int n4) {
  int i = blockIdx.x * blockDim.x + threadIdx.x;
  if (i < n4) {
    float4 v = in[i];
    ushort4 o;
    o.x = f2bf(v.x); o.y = f2bf(v.y); o.z = f2bf(v.z); o.w = f2bf(v.w);
    out[i] = o;
  }
}

// in [rows][cols] f32 -> out [cols][rows] bf16
__global__ void cvtT_bf16_kernel(const float* __restrict__ in, u16* __restrict__ out,
                                 int rows, int cols) {
  int i = blockIdx.x * blockDim.x + threadIdx.x;
  if (i < rows * cols) {
    int r = i / cols, c = i % cols;
    out[(size_t)c * rows + r] = f2bf(in[i]);
  }
}

// ================= LDS-staged double-buffered GEMM core =================
// Block: 256 thr (8 waves). Block tile 128(M) x 128(N), K-step 64.
// Wave (wv): rows (wv>>1)*32..+31, cols (wv&1)*64..+63 -> 16 WMMAs / K-step.
// Staging: A tile 128x64 bf16 (16KB) + B tile 128x64 (16KB), double buffered
// (64KB LDS). Each thread moves one 64B half-row of A and of B per stage.

#define GEMM_BODY()                                                                    \
  const int lane = threadIdx.x & 31;                                                   \
  const int wv   = threadIdx.x >> 5;                                                   \
  const int tid  = threadIdx.x;                                                        \
  const int mw = (wv >> 1) * 32;                                                       \
  const int nw = (wv & 1) * 64;                                                        \
  const int m0 = blockIdx.x * 128;                                                     \
  const int n0 = blockIdx.y * 128;                                                     \
  __shared__ __align__(16) u16 shA[2][128 * 64];                                       \
  __shared__ __align__(16) u16 shB[2][128 * 64];                                       \
  const int r0  = tid >> 1;           /* 0..127: one row of A and one of B */          \
  const int c32 = (tid & 1) << 5;     /* 0 or 32: half-row */                          \
  const u16* gA = Abase + (size_t)(m0 + r0) * N_EMBD + c32;                            \
  const u16* gB = Bbase + (size_t)(n0 + r0) * N_EMBD + c32;                            \
  const int ls = r0 * 64 + c32;                                                        \
  v8f c00 = {}, c01 = {}, c02 = {}, c03 = {};                                          \
  v8f c10 = {}, c11 = {}, c12 = {}, c13 = {};                                          \
  const int NK = N_EMBD / 64;

// one K=32 sub-step on named accumulators (no array indirection)
#define GEMM_COMPUTE_STEP(sA, sB, kb)                                                  \
  {                                                                                    \
    v16bf a0 = load_frag_a((sA), mw,      64, (kb), lane);                             \
    v16bf a1 = load_frag_a((sA), mw + 16, 64, (kb), lane);                             \
    v16bf b0 = load_frag_b((sB), nw +  0, 64, (kb), lane);                             \
    v16bf b1 = load_frag_b((sB), nw + 16, 64, (kb), lane);                             \
    v16bf b2 = load_frag_b((sB), nw + 32, 64, (kb), lane);                             \
    v16bf b3 = load_frag_b((sB), nw + 48, 64, (kb), lane);                             \
    c00 = wmma_bf16(a0, b0, c00);  c10 = wmma_bf16(a1, b0, c10);                       \
    c01 = wmma_bf16(a0, b1, c01);  c11 = wmma_bf16(a1, b1, c11);                       \
    c02 = wmma_bf16(a0, b2, c02);  c12 = wmma_bf16(a1, b2, c12);                       \
    c03 = wmma_bf16(a0, b3, c03);  c13 = wmma_bf16(a1, b3, c13);                       \
  }

#define GEMM_COMPUTE(p)                                                                \
  GEMM_COMPUTE_STEP(&shA[p][0], &shB[p][0], 0)                                         \
  GEMM_COMPUTE_STEP(&shA[p][0], &shB[p][0], 32)

#if HAVE_ASYNC_LDS
// single barrier / iteration: wait own async, barrier, kick next stage, compute.
#define GEMM_MAINLOOP()                                                                \
  {                                                                                    \
    auto issue = [&](int kt, int p) {                                                  \
      const size_t ko = (size_t)kt * 64;                                               \
      _Pragma("unroll")                                                                \
      for (int i = 0; i < 4; ++i) {                                                    \
        __builtin_amdgcn_global_load_async_to_lds_b128(                                \
            AS_GLOBAL_V4I(gA + ko + i * 8), AS_LOCAL_V4I(&shA[p][ls + i * 8]), 0, 0);  \
        __builtin_amdgcn_global_load_async_to_lds_b128(                                \
            AS_GLOBAL_V4I(gB + ko + i * 8), AS_LOCAL_V4I(&shB[p][ls + i * 8]), 0, 0);  \
      }                                                                                \
    };                                                                                 \
    issue(0, 0);                                                                       \
    for (int kt = 0; kt < NK; ++kt) {                                                  \
      const int p = kt & 1;                                                            \
      wait_asynccnt_0();                                                               \
      __syncthreads();                                                                 \
      if (kt + 1 < NK) issue(kt + 1, p ^ 1);                                           \
      GEMM_COMPUTE(p)                                                                  \
    }                                                                                  \
  }
#else
#define GEMM_MAINLOOP()                                                                \
  { /* register-relay double-buffered pipeline */                                      \
    uint4 tA[4], tB[4];                                                                \
    auto gload = [&](int kt) {                                                         \
      const size_t ko = (size_t)kt * 64;                                               \
      _Pragma("unroll")                                                                \
      for (int i = 0; i < 4; ++i) {                                                    \
        tA[i] = *(const uint4*)(gA + ko + i * 8);                                      \
        tB[i] = *(const uint4*)(gB + ko + i * 8);                                      \
      }                                                                                \
    };                                                                                 \
    auto lstore = [&](int p) {                                                         \
      _Pragma("unroll")                                                                \
      for (int i = 0; i < 4; ++i) {                                                    \
        *(uint4*)&shA[p][ls + i * 8] = tA[i];                                          \
        *(uint4*)&shB[p][ls + i * 8] = tB[i];                                          \
      }                                                                                \
    };                                                                                 \
    gload(0); lstore(0);                                                               \
    for (int kt = 0; kt < NK; ++kt) {                                                  \
      const int p = kt & 1;                                                            \
      __syncthreads();                                                                 \
      if (kt + 1 < NK) gload(kt + 1);                                                  \
      GEMM_COMPUTE(p)                                                                  \
      __syncthreads();                                                                 \
      if (kt + 1 < NK) lstore(p ^ 1);                                                  \
    }                                                                                  \
  }
#endif

// ---------------- QKV GEMM: xb[8192,1024] @ WqkvT[3072,1024]^T ----------------

__global__ __launch_bounds__(256) void qkv_gemm_kernel(
    const u16* __restrict__ Abase, const u16* __restrict__ Bbase,
    u16* __restrict__ qb, u16* __restrict__ kb, u16* __restrict__ vtb) {
  GEMM_BODY()
  GEMM_MAINLOOP()

  v8f acc[2][4] = {{c00, c01, c02, c03}, {c10, c11, c12, c13}};
  const int half8 = (lane >> 4) << 3;
#pragma unroll
  for (int mi = 0; mi < 2; ++mi) {
#pragma unroll
    for (int j = 0; j < 4; ++j) {
      const int n = n0 + nw + j * 16 + (lane & 15);
      const int which = n >> 10;          // 0=q 1=k 2=v
      const int c = n & (N_EMBD - 1);
      const int h = c >> 6;
      const int d = c & (DH - 1);
#pragma unroll
      for (int r = 0; r < 8; ++r) {
        const int m = m0 + mw + mi * 16 + r + half8;
        const int b = m >> 11;            // m / 2048
        const int t = m & (TSEQ - 1);
        const u16 bv = f2bf(acc[mi][j][r]);
        const size_t bh = (size_t)(b * N_HEAD + h);
        if (which == 0)       qb [(bh * TSEQ + t) * DH + d] = bv;
        else if (which == 1)  kb [(bh * TSEQ + t) * DH + d] = bv;
        else                  vtb[(bh * DH + d) * TSEQ + t] = bv;   // V^T [Dh][T]
      }
    }
  }
}

// ---------------- output projection: yb[8192,1024] @ WprojT[1024,1024]^T -> f32 ----

__global__ __launch_bounds__(256) void proj_gemm_kernel(
    const u16* __restrict__ Abase, const u16* __restrict__ Bbase,
    float* __restrict__ out) {
  GEMM_BODY()
  GEMM_MAINLOOP()

  v8f acc[2][4] = {{c00, c01, c02, c03}, {c10, c11, c12, c13}};
  const int half8 = (lane >> 4) << 3;
#pragma unroll
  for (int mi = 0; mi < 2; ++mi) {
#pragma unroll
    for (int j = 0; j < 4; ++j) {
      const int n = n0 + nw + j * 16 + (lane & 15);
#pragma unroll
      for (int r = 0; r < 8; ++r) {
        const int m = m0 + mw + mi * 16 + r + half8;
        out[(size_t)m * N_EMBD + n] = acc[mi][j][r];
      }
    }
  }
}

// ---------------- causal flash attention ----------------
// block = 128 thr (4 waves); block owns (b,h, 64 q-rows); wave owns 16 q-rows.

__global__ __launch_bounds__(128) void attn_kernel(
    const u16* __restrict__ qb, const u16* __restrict__ kb,
    const u16* __restrict__ vtb, u16* __restrict__ yb) {
  const int lane = threadIdx.x & 31;
  const int wv   = threadIdx.x >> 5;
  const int qt = blockIdx.x & 31;           // 32 q-tiles of 64
  const int bh = blockIdx.x >> 5;           // 0..63
  const int h  = bh & (N_HEAD - 1);
  const int b  = bh >> 4;

  const u16* qh  = qb  + (size_t)bh * TSEQ * DH;
  const u16* kh  = kb  + (size_t)bh * TSEQ * DH;
  const u16* vth = vtb + (size_t)bh * DH * TSEQ;

  const int qr0   = qt * 64 + wv * 16;
  const int half8 = (lane >> 4) << 3;
  const int ncol  = lane & 15;

  // q fragments for Dh=64 (two K=32 slices), resident in registers
  const v16bf qa0 = load_frag_a(qh, qr0, DH, 0, lane);
  const v16bf qa1 = load_frag_a(qh, qr0, DH, 32, lane);

  v8f o[4] = {};
  float mrow[8], lrow[8];
#pragma unroll
  for (int r = 0; r < 8; ++r) { mrow[r] = -1e30f; lrow[r] = 0.0f; }

  __shared__ __align__(16) u16 pshm[4][16 * 32];
  u16* pw = &pshm[wv][0];

  const int kend = qr0 + 15;                // last key column needed (causal)
  for (int kc = 0; kc <= kend; kc += 32) {
    if (kc + 32 <= kend) {                  // near-scope prefetch of next K/V tiles
      __builtin_prefetch(kh + (size_t)(kc + 32 + ncol) * DH, 0, 3);
      __builtin_prefetch(vth + (size_t)ncol * TSEQ + kc + 32, 0, 3);
    }

    // ---- scores S = Q @ K^T (16 x 32), two 16x16 C tiles ----
    v8f s0 = {}, s1 = {};
    {
      v16bf b00 = load_frag_b(kh, kc,      DH, 0,  lane);
      v16bf b01 = load_frag_b(kh, kc,      DH, 32, lane);
      v16bf b10 = load_frag_b(kh, kc + 16, DH, 0,  lane);
      v16bf b11 = load_frag_b(kh, kc + 16, DH, 32, lane);
      s0 = wmma_bf16(qa0, b00, s0);
      s0 = wmma_bf16(qa1, b01, s0);
      s1 = wmma_bf16(qa0, b10, s1);
      s1 = wmma_bf16(qa1, b11, s1);
    }

    // ---- V fragments issued early: their loads overlap the softmax VALU ----
    v16bf vf[4];
#pragma unroll
    for (int a = 0; a < 4; ++a)
      vf[a] = load_frag_b(vth, a * 16, TSEQ, kc, lane);

    // ---- online softmax (rows live across 16-lane groups) ----
#pragma unroll
    for (int r = 0; r < 8; ++r) {
      const int row = qr0 + r + half8;
      float x0 = s0[r] * INV_SCALE;
      float x1 = s1[r] * INV_SCALE;
      if (kc + ncol      > row) x0 = -1e30f;     // causal mask
      if (kc + 16 + ncol > row) x1 = -1e30f;

      float cand = fmaxf(x0, x1);
#pragma unroll
      for (int off = 1; off < 16; off <<= 1)
        cand = fmaxf(cand, __shfl_xor(cand, off, 32));
      const float mnew = fmaxf(mrow[r], cand);
      const float scl  = __expf(mrow[r] - mnew);
      const float p0 = __expf(x0 - mnew);
      const float p1 = __expf(x1 - mnew);
      float psum = p0 + p1;
#pragma unroll
      for (int off = 1; off < 16; off <<= 1)
        psum += __shfl_xor(psum, off, 32);
      lrow[r] = lrow[r] * scl + psum;
      mrow[r] = mnew;
      o[0][r] *= scl; o[1][r] *= scl; o[2][r] *= scl; o[3][r] *= scl;

      pw[(r + half8) * 32 + ncol]      = f2bf(p0);
      pw[(r + half8) * 32 + 16 + ncol] = f2bf(p1);
    }

    // wave-private LDS tile: wait for our ds stores, then reload in A layout
    asm volatile("s_wait_dscnt 0" ::: "memory");
    const v16bf pa = load_frag_a(pw, 0, 32, 0, lane);

    // ---- O = O*diag(scl) + P @ V ----
#pragma unroll
    for (int a = 0; a < 4; ++a)
      o[a] = wmma_bf16(pa, vf[a], o[a]);
  }

  // ---- epilogue: y[b][t][h*64 + d] = O / l ----
#pragma unroll
  for (int r = 0; r < 8; ++r) {
    const int row = qr0 + r + half8;
    const float inv = 1.0f / lrow[r];
#pragma unroll
    for (int a = 0; a < 4; ++a) {
      const int c = h * DH + a * 16 + ncol;
      yb[((size_t)b * TSEQ + row) * N_EMBD + c] = f2bf(o[a][r] * inv);
    }
  }
}

// ---------------- host launch ----------------

extern "C" void kernel_launch(void* const* d_in, const int* in_sizes, int n_in,
                              void* d_out, int out_size, void* d_ws, size_t ws_size,
                              hipStream_t stream) {
  (void)in_sizes; (void)n_in; (void)out_size; (void)ws_size;
  const float* x     = (const float*)d_in[0];
  const float* Wqkv  = (const float*)d_in[1];
  const float* Wproj = (const float*)d_in[2];
  float* out = (float*)d_out;

  u16* ws = (u16*)d_ws;
  size_t off = 0;
  u16* xb     = ws + off; off += (size_t)MROWS * N_EMBD;             // x bf16 [8192,1024]
  u16* wqkvT  = ws + off; off += (size_t)QKV_N * N_EMBD;             // Wqkv^T bf16 [3072,1024]
  u16* wprojT = ws + off; off += (size_t)N_EMBD * N_EMBD;            // Wproj^T bf16 [1024,1024]
  u16* qb     = ws + off; off += (size_t)BATCH * N_HEAD * TSEQ * DH; // q [B,H,T,Dh]
  u16* kb     = ws + off; off += (size_t)BATCH * N_HEAD * TSEQ * DH; // k [B,H,T,Dh]
  u16* vtb    = ws + off; off += (size_t)BATCH * N_HEAD * DH * TSEQ; // v^T [B,H,Dh,T]
  u16* yb     = ws + off; off += (size_t)MROWS * N_EMBD;             // y bf16 [8192,1024]

  {
    int n4 = (MROWS * N_EMBD) / 4;
    cvt_bf16x4_kernel<<<(n4 + 255) / 256, 256, 0, stream>>>(
        (const float4*)x, (ushort4*)xb, n4);
  }
  {
    int n = N_EMBD * QKV_N;
    cvtT_bf16_kernel<<<(n + 255) / 256, 256, 0, stream>>>(Wqkv, wqkvT, N_EMBD, QKV_N);
  }
  {
    int n = N_EMBD * N_EMBD;
    cvtT_bf16_kernel<<<(n + 255) / 256, 256, 0, stream>>>(Wproj, wprojT, N_EMBD, N_EMBD);
  }

  qkv_gemm_kernel<<<dim3(MROWS / 128, QKV_N / 128), 256, 0, stream>>>(xb, wqkvT, qb, kb, vtb);
  attn_kernel<<<BATCH * N_HEAD * (TSEQ / 64), 128, 0, stream>>>(qb, kb, vtb, yb);
  proj_gemm_kernel<<<dim3(MROWS / 128, N_EMBD / 128), 256, 0, stream>>>(yb, wprojT, out);
}
